// RNNImputation_37649683316768
// MI455X (gfx1250) — matrix operand order repaired and checked by reading, not compiled, not run on hardware
//
#include <hip/hip_runtime.h>

// GRU-with-imputation recurrence for MI455X (gfx1250, wave32, WMMA).
// B=256 batch rows -> 16 workgroups x 16 rows; each WG runs its 2048-step
// chain with Whh resident in VGPRs as f16 B-fragments and h resident in
// registers (f32), sharing h across waves via LDS in WMMA A-fragment layout.
// Gate nonlinearities use hw trans ops (v_exp_f32 / v_rcp_f32 / v_tanh_f32).
// The t-loop is forced to unroll(1) and the f16 weight fragments are pinned
// with empty inline asm so the compiler keeps them resident instead of
// re-converting from f32 every iteration (observed in round-2 codegen).

#define B_TOT   256
#define T_TOT   2048
#define HID     128
#define MISSING_V 128.0f
#define LOG2E   1.4426950408889634f

typedef _Float16 v16h __attribute__((ext_vector_type(16)));
typedef _Float16 v8h  __attribute__((ext_vector_type(8)));
typedef float    v8f  __attribute__((ext_vector_type(8)));

// sigmoid(logit) where pre == -log2(e) * logit  ->  rcp(1 + exp2(pre))
__device__ __forceinline__ float sigmoid_from_pre(float pre) {
    const float e = __builtin_amdgcn_exp2f(pre);
    return __builtin_amdgcn_rcpf(1.0f + e);
}

__device__ __forceinline__ float fast_tanh(float u) {
#if __has_builtin(__builtin_amdgcn_tanhf)
    return __builtin_amdgcn_tanhf(u);            // v_tanh_f32 (gfx1250 trans op)
#elif __has_builtin(__builtin_amdgcn_tanh_f32)
    return __builtin_amdgcn_tanh_f32(u);
#else
    // tanh(u) = 1 - 2 / (1 + e^{2u})
    const float e = __builtin_amdgcn_exp2f(u * (2.0f * LOG2E));
    return __builtin_fmaf(-2.0f, __builtin_amdgcn_rcpf(1.0f + e), 1.0f);
#endif
}

__global__ __launch_bounds__(256, 1)
void RNNImputation_gru_wmma_kernel(const float* __restrict__ x,    // [B, T, 1]
                                   const float* __restrict__ Wih,  // [3H, 1]
                                   const float* __restrict__ Whh,  // [3H, H]
                                   const float* __restrict__ bih,  // [3H]
                                   const float* __restrict__ bhh,  // [3H]
                                   const float* __restrict__ Wfc,  // [1, H]
                                   const float* __restrict__ bfc,  // [1]
                                   float* __restrict__ out_new,    // [T, B]
                                   float* __restrict__ out_pred)   // [B*(T-1)]
{
    // Double-buffered hidden state in WMMA A-source layout: [batch row m][hidden k]
    __shared__ __align__(16) _Float16 h16[2][16][HID];
    __shared__ float curs[16];          // imputed input per batch row this step
    __shared__ float xpart[8][16];      // per-wave partial sums of h . Wfc

    const int tid    = threadIdx.x;
    const int wav    = tid >> 5;        // 0..7, owns hidden cols [16w,16w+16)
    const int lane   = tid & 31;
    const int n      = lane & 15;       // tile column
    const int hiHalf = lane >> 4;       // lane half selects K sub-block
    const int c      = (wav << 4) + n;  // hidden unit owned by this lane
    const int bbase  = blockIdx.x * 16; // first batch row of this WG

    // ---- per-lane gate constants (I == 1 makes gi rank-1 / elementwise) ----
    const float wihR = Wih[c];
    const float wihZ = Wih[HID + c];
    const float wihN = Wih[2*HID + c];
    const float bR   = bih[c]         + bhh[c];
    const float bZ   = bih[HID + c]   + bhh[HID + c];
    const float biN  = bih[2*HID + c];               // bhh_n multiplies by r
    const float bhN  = bhh[2*HID + c];
    const float wfc  = Wfc[c];
    const float bfcv = bfc[0];
    // pre-scaled by -log2(e) so each sigmoid is 2 FMA + exp2 + add + rcp
    const float wihRs = -LOG2E * wihR, bRs = -LOG2E * bR;
    const float wihZs = -LOG2E * wihZ, bZs = -LOG2E * bZ;

    // ---- Whh as f16 B-fragments, resident in VGPRs for the whole chain ----
    // B (32x16) layout: lane = column n (+16 for K-half), element e <-> k = 16*hiHalf + e
    v16h fragR[4], fragZ[4], fragN[4];
#pragma unroll
    for (int kc = 0; kc < 4; ++kc) {
        const int kb = kc * 32 + hiHalf * 16;
        const float* wr = Whh + (size_t)c * HID + kb;
        const float* wz = Whh + (size_t)(HID   + c) * HID + kb;
        const float* wn = Whh + (size_t)(2*HID + c) * HID + kb;
#pragma unroll
        for (int e = 0; e < 16; ++e) {
            fragR[kc][e] = (_Float16)wr[e];
            fragZ[kc][e] = (_Float16)wz[e];
            fragN[kc][e] = (_Float16)wn[e];
        }
        // Pin the converted f16 tuples in VGPRs: the empty asm makes them opaque
        // so the backend cannot keep the f32 sources live and re-convert per step.
        asm volatile("" : "+v"(fragR[kc]));
        asm volatile("" : "+v"(fragZ[kc]));
        asm volatile("" : "+v"(fragN[kc]));
    }

    // ---- zero initial state ----
    for (int i = tid; i < 16 * HID; i += blockDim.x)
        ((_Float16*)h16[0])[i] = (_Float16)0.0f;
    for (int i = tid; i < 8 * 16; i += blockDim.x)
        ((float*)xpart)[i] = 0.0f;

    float hprev[8];                      // h[m = v + 8*hiHalf][c], f32 master copy
#pragma unroll
    for (int v = 0; v < 8; ++v) hprev[v] = 0.0f;

    // software-pipelined x load (wave 0 front lanes drive the input path)
    float xv = 0.0f;
    if (tid < 16) xv = x[(size_t)(bbase + tid) * T_TOT + 0];
    __syncthreads();

#pragma unroll 1   // strictly serial recurrence: unrolling only bloats registers
    for (int t = 0; t < T_TOT; ++t) {
        const int pb = t & 1;

        // -------- phase 1: imputation + outputs (wave 0, lanes 0..15) --------
        if (tid < 16) {
            const int gb = bbase + tid;
            float xh = bfcv;                  // x_hat = h_{t-1} . Wfc + bfc
#pragma unroll
            for (int w = 0; w < 8; ++w) xh += xpart[w][tid];

            float cur;
            if (t == 0) {
                cur = xv;                     // step 0: no imputation
            } else {
                cur = (xv == MISSING_V) ? xh : xv;
                out_pred[(size_t)gb * (T_TOT - 1) + (t - 1)] = xh;
            }
            out_new[(size_t)t * B_TOT + gb] = cur;
            curs[tid] = cur;

            // pipeline next x one step ahead (hides L2/HBM latency)
            if (t + 1 < T_TOT) {
                xv = x[(size_t)gb * T_TOT + t + 1];
                if (t + 33 < T_TOT)
                    __builtin_prefetch(&x[(size_t)gb * T_TOT + t + 33], 0, 1);
            }
        }
        __syncthreads();   // cur ready; previous step's xpart consumed

        // -------- phase 2: gh = h . Whh^T via v_wmma_f32_16x16x32_f16 --------
        v8f aR = {}; v8f aZ = {}; v8f aN = {};
#pragma unroll
        for (int kc = 0; kc < 4; ++kc) {
            // A (16x32) layout: lane m = n; element e <-> K = 16*hiHalf + e
            const _Float16* row = &h16[pb][n][kc * 32 + hiHalf * 16];
            v8h lo = *(const v8h*)row;
            v8h hi = *(const v8h*)(row + 8);
            v16h afrag = __builtin_shufflevector(lo, hi,
                0,1,2,3,4,5,6,7,8,9,10,11,12,13,14,15);
            aR = __builtin_amdgcn_wmma_f32_16x16x32_f16(false, afrag, false, fragR[kc],
                                                        (short)0, aR, false, false);
            aZ = __builtin_amdgcn_wmma_f32_16x16x32_f16(false, afrag, false, fragZ[kc],
                                                        (short)0, aZ, false, false);
            aN = __builtin_amdgcn_wmma_f32_16x16x32_f16(false, afrag, false, fragN[kc],
                                                        (short)0, aN, false, false);
        }

        // -------- gate math + state update (D layout: VGPR v -> row v / v+8) --------
        const int mofs = hiHalf * 8;
        float xp[8];
#pragma unroll
        for (int v = 0; v < 8; ++v) {
            const int   m  = v + mofs;
            const float cu = curs[m];
            const float r  = sigmoid_from_pre(
                __builtin_fmaf(aR[v], -LOG2E, __builtin_fmaf(cu, wihRs, bRs)));
            const float z  = sigmoid_from_pre(
                __builtin_fmaf(aZ[v], -LOG2E, __builtin_fmaf(cu, wihZs, bZs)));
            const float u  = __builtin_fmaf(r, aN[v] + bhN,
                             __builtin_fmaf(cu, wihN, biN));
            const float g  = fast_tanh(u);
            const float h  = __builtin_fmaf(1.0f - z, g, z * hprev[v]);
            hprev[v] = h;
            h16[pb ^ 1][m][c] = (_Float16)h;   // publish f16 copy for next step's A
            xp[v] = h * wfc;                    // contribution to x_hat[m]
        }

        // deterministic cross-lane reduction of x_hat partials (16 lanes share a row)
#pragma unroll
        for (int v = 0; v < 8; ++v) {
            float s = xp[v];
            s += __shfl_xor(s, 1);
            s += __shfl_xor(s, 2);
            s += __shfl_xor(s, 4);
            s += __shfl_xor(s, 8);
            if ((lane & 15) == 0)
                xpart[wav][v + mofs] = s;      // fixed writer per slot -> deterministic
        }
        __syncthreads();   // h16[pb^1] + xpart complete before next step
    }
}

extern "C" void kernel_launch(void* const* d_in, const int* in_sizes, int n_in,
                              void* d_out, int out_size, void* d_ws, size_t ws_size,
                              hipStream_t stream) {
    (void)in_sizes; (void)n_in; (void)out_size; (void)d_ws; (void)ws_size;
    const float* x   = (const float*)d_in[0];
    const float* Wih = (const float*)d_in[1];
    const float* Whh = (const float*)d_in[2];
    const float* bih = (const float*)d_in[3];
    const float* bhh = (const float*)d_in[4];
    const float* Wfc = (const float*)d_in[5];
    const float* bfc = (const float*)d_in[6];

    float* out_new  = (float*)d_out;                       // [T, B] = 524288 f32
    float* out_pred = out_new + (size_t)T_TOT * B_TOT;     // [B*(T-1)] = 524032 f32

    RNNImputation_gru_wmma_kernel<<<dim3(B_TOT / 16), dim3(256), 0, stream>>>(
        x, Wih, Whh, bih, bhh, Wfc, bfc, out_new, out_pred);
}
